// YangAttnetion_71210557767777
// MI455X (gfx1250) — matrix operand order repaired and checked by reading, not compiled, not run on hardware
//
#include <hip/hip_runtime.h>

typedef __attribute__((ext_vector_type(16))) _Float16 v16h;
typedef __attribute__((ext_vector_type(8)))  _Float16 h8;
typedef __attribute__((ext_vector_type(8)))  float    v8f;
typedef __attribute__((ext_vector_type(4)))  float    f4;

#define SEQ     2048
#define DIM     256
#define BATCH   64
#define M_TOTAL (BATCH * SEQ)   // 131072
#define MB      128             // M rows per block
#define KC      32              // K chunk (one f16-WMMA depth)
#define LDH     40              // padded halfs per LDS row (80B: 16B-aligned, bank-spread)
#define SCHUNK  256             // S elements per pass-2b block
#define NCHUNK  (SEQ / SCHUNK)  // 8

__device__ __forceinline__ float fast_tanh(float x) {
    // tanh(x) = 1 - 2/(exp(2x)+1); v_exp_f32 + v_rcp_f32, exact at saturation
    float e = __expf(2.0f * x);
    return 1.0f - 2.0f * __builtin_amdgcn_rcpf(e + 1.0f);
}

__device__ __forceinline__ h8 cvt_f32x8_to_h8(f4 a, f4 b) {
    h8 r;
    r[0] = (_Float16)a[0]; r[1] = (_Float16)a[1];
    r[2] = (_Float16)a[2]; r[3] = (_Float16)a[3];
    r[4] = (_Float16)b[0]; r[5] = (_Float16)b[1];
    r[6] = (_Float16)b[2]; r[7] = (_Float16)b[3];
    return r;
}

// ---------------------------------------------------------------------------
// Pass 1: scores[m] = ctx . tanh( X[m,:] * W^T + b )   for m = b*SEQ + s
// Block: 128 M-rows x 256 N, 8 waves; wave tile = 64M x 64N (4x4 WMMA accs).
// ---------------------------------------------------------------------------
__global__ __launch_bounds__(256)
void pass1_scores(const float* __restrict__ X, const float* __restrict__ W,
                  const float* __restrict__ bvec, const float* __restrict__ ctx,
                  float* __restrict__ scores)
{
    __shared__ _Float16 Xs[MB * LDH];     // 10.0 KB
    __shared__ _Float16 Ws[DIM * LDH];    // 20.0 KB
    __shared__ float    sredp[4][MB];     // per-(n-wave) score partials, 2 KB

    const int t    = threadIdx.x;
    const int m0   = blockIdx.x * MB;
    const int lane = t & 31;
    const int wid  = t >> 5;
    const int wm   = wid >> 2;           // 0..1 : M offset = 64*wm
    const int wn   = wid & 3;            // 0..3 : N offset = 64*wn
    const int half = lane >> 4;          // 0|1
    const int l16  = lane & 15;

    v8f acc[4][4];
#pragma unroll
    for (int i = 0; i < 4; ++i)
#pragma unroll
        for (int j = 0; j < 4; ++j) acc[i][j] = (v8f)0.0f;

    const int xrow = t >> 1, xseg = (t & 1) * 16;
    const float* xsrc_base = X + (size_t)(m0 + xrow) * DIM + xseg;
    const float* wsrc_base = W + (size_t)t * DIM;

    for (int kc = 0; kc < DIM; kc += KC) {
        __syncthreads();
        { // stage X slab: thread -> (row t/2, 16-float segment), f32 -> f16
            const float* src = xsrc_base + kc;
            f4 a0 = *(const f4*)(src + 0);
            f4 a1 = *(const f4*)(src + 4);
            f4 a2 = *(const f4*)(src + 8);
            f4 a3 = *(const f4*)(src + 12);
            *(h8*)&Xs[xrow * LDH + xseg    ] = cvt_f32x8_to_h8(a0, a1);
            *(h8*)&Xs[xrow * LDH + xseg + 8] = cvt_f32x8_to_h8(a2, a3);
        }
        { // stage W slab: thread t -> W row t (n=t), 32 floats -> 32 halfs
            const float* src = wsrc_base + kc;
            f4 a0 = *(const f4*)(src +  0);
            f4 a1 = *(const f4*)(src +  4);
            f4 a2 = *(const f4*)(src +  8);
            f4 a3 = *(const f4*)(src + 12);
            f4 a4 = *(const f4*)(src + 16);
            f4 a5 = *(const f4*)(src + 20);
            f4 a6 = *(const f4*)(src + 24);
            f4 a7 = *(const f4*)(src + 28);
            *(h8*)&Ws[t * LDH +  0] = cvt_f32x8_to_h8(a0, a1);
            *(h8*)&Ws[t * LDH +  8] = cvt_f32x8_to_h8(a2, a3);
            *(h8*)&Ws[t * LDH + 16] = cvt_f32x8_to_h8(a4, a5);
            *(h8*)&Ws[t * LDH + 24] = cvt_f32x8_to_h8(a6, a7);
        }
        // prefetch next chunk (global_prefetch_b8) to overlap with WMMA phase
        if (kc + KC < DIM) {
            __builtin_prefetch(xsrc_base + kc + KC, 0, 3);
            __builtin_prefetch(wsrc_base + kc + KC, 0, 3);
        }
        __syncthreads();

        // A fragments: 16x32 f16 per ISA layout (lane<16: K 0-7,16-23; lane>=16: +8)
        v16h afrag[4];
#pragma unroll
        for (int mi = 0; mi < 4; ++mi) {
            int row = wm * 64 + mi * 16 + l16;
            h8 lo = *(const h8*)&Xs[row * LDH + half * 8];
            h8 hi = *(const h8*)&Xs[row * LDH + 16 + half * 8];
            afrag[mi] = __builtin_shufflevector(lo, hi,
                0, 1, 2, 3, 4, 5, 6, 7, 8, 9, 10, 11, 12, 13, 14, 15);
        }
#pragma unroll
        for (int ni = 0; ni < 4; ++ni) {
            // B fragment: column n, K 0-7,16-23 (lane<16) / +8 (lane>=16), matching A
            int col = wn * 64 + ni * 16 + l16;
            h8 blo = *(const h8*)&Ws[col * LDH + half * 8];
            h8 bhi = *(const h8*)&Ws[col * LDH + 16 + half * 8];
            v16h bfrag = __builtin_shufflevector(blo, bhi,
                0, 1, 2, 3, 4, 5, 6, 7, 8, 9, 10, 11, 12, 13, 14, 15);
#pragma unroll
            for (int mi = 0; mi < 4; ++mi) {
                acc[mi][ni] = __builtin_amdgcn_wmma_f32_16x16x32_f16(
                    false, afrag[mi], false, bfrag,
                    (short)0, acc[mi][ni], false, false);
            }
        }
    }

    // Epilogue: u = tanh(acc + b[n]); partial score += ctx[n] * u  (u never stored)
    float sc[4][8];
#pragma unroll
    for (int mi = 0; mi < 4; ++mi)
#pragma unroll
        for (int r = 0; r < 8; ++r) sc[mi][r] = 0.0f;

#pragma unroll
    for (int ni = 0; ni < 4; ++ni) {
        int n_g = wn * 64 + ni * 16 + l16;
        float bb = bvec[n_g];
        float cc = ctx[n_g];
#pragma unroll
        for (int mi = 0; mi < 4; ++mi)
#pragma unroll
            for (int r = 0; r < 8; ++r)
                sc[mi][r] += cc * fast_tanh(acc[mi][ni][r] + bb);
    }

    // reduce across the 16-lane halves (xor 1,2,4,8 stays within each half)
#pragma unroll
    for (int mi = 0; mi < 4; ++mi) {
#pragma unroll
        for (int r = 0; r < 8; ++r) {
            float v = sc[mi][r];
            v += __shfl_xor(v, 1, 32);
            v += __shfl_xor(v, 2, 32);
            v += __shfl_xor(v, 4, 32);
            v += __shfl_xor(v, 8, 32);
            if (l16 == 0)  // lanes 0 and 16 own m_local = ..+0 / ..+8
                sredp[wn][wm * 64 + mi * 16 + r + 8 * half] = v;
        }
    }
    __syncthreads();
    // deterministic fixed-order combine of the 4 N-waves
    if (t < MB)
        scores[m0 + t] = (sredp[0][t] + sredp[1][t]) + (sredp[2][t] + sredp[3][t]);
}

// ---------------------------------------------------------------------------
// Pass 2a: per-batch softmax; normalizes weights in place in the ws buffer
// ---------------------------------------------------------------------------
__global__ __launch_bounds__(256)
void pass2a_softmax(float* __restrict__ scores)
{
    __shared__ float red[256];
    const int b = blockIdx.x, t = threadIdx.x;
    float* sc = scores + (size_t)b * SEQ;

    float vloc[8];
    float mx = -1e30f;
#pragma unroll
    for (int i = 0; i < 8; ++i) {
        vloc[i] = sc[t + 256 * i];
        mx = fmaxf(mx, vloc[i]);
    }
    red[t] = mx;
    __syncthreads();
    for (int o = 128; o > 0; o >>= 1) {
        if (t < o) red[t] = fmaxf(red[t], red[t + o]);
        __syncthreads();
    }
    mx = red[0];
    __syncthreads();

    float sum = 0.0f;
#pragma unroll
    for (int i = 0; i < 8; ++i) {
        vloc[i] = __expf(vloc[i] - mx);
        sum += vloc[i];
    }
    red[t] = sum;
    __syncthreads();
    for (int o = 128; o > 0; o >>= 1) {
        if (t < o) red[t] += red[t + o];
        __syncthreads();
    }
    float inv = 1.0f / red[0];
#pragma unroll
    for (int i = 0; i < 8; ++i)
        sc[t + 256 * i] = vloc[i] * inv;
}

// ---------------------------------------------------------------------------
// Pass 2b: partial[b, c, d] = sum_{s in chunk c} w[b,s] * x[b,s,d]
// 512 blocks for occupancy on the 128 MiB stream (L2-resident from pass 1)
// ---------------------------------------------------------------------------
__global__ __launch_bounds__(256)
void pass2b_partial(const float* __restrict__ X, const float* __restrict__ wgt,
                    float* __restrict__ partial)
{
    __shared__ float wl[SCHUNK];
    const int blk = blockIdx.x;
    const int b = blk >> 3, c = blk & (NCHUNK - 1);
    const int t = threadIdx.x;
    const int s0 = c * SCHUNK;

    wl[t] = wgt[(size_t)b * SEQ + s0 + t];
    __syncthreads();

    const float* xb = X + ((size_t)b * SEQ + s0) * DIM;
    float acc = 0.0f;
#pragma unroll 8
    for (int s = 0; s < SCHUNK; ++s)
        acc = fmaf(wl[s], xb[(size_t)s * DIM + t], acc);
    partial[(size_t)blk * DIM + t] = acc;
}

// ---------------------------------------------------------------------------
// Pass 2c: out[b,d] = sum_c partial[b,c,d]   (deterministic fixed order)
// ---------------------------------------------------------------------------
__global__ __launch_bounds__(256)
void pass2c_reduce(const float* __restrict__ partial, float* __restrict__ out)
{
    const int b = blockIdx.x, t = threadIdx.x;
    const float* p = partial + (size_t)b * NCHUNK * DIM + t;
    float a = 0.0f;
#pragma unroll
    for (int c = 0; c < NCHUNK; ++c)
        a += p[c * DIM];
    out[b * DIM + t] = a;
}

extern "C" void kernel_launch(void* const* d_in, const int* in_sizes, int n_in,
                              void* d_out, int out_size, void* d_ws, size_t ws_size,
                              hipStream_t stream) {
    const float* X    = (const float*)d_in[0];   // [64, 2048, 256]
    const float* W    = (const float*)d_in[1];   // [256, 256]
    const float* bvec = (const float*)d_in[2];   // [256]
    const float* ctx  = (const float*)d_in[3];   // [256]
    float* out     = (float*)d_out;              // [1, 64, 256]
    float* scores  = (float*)d_ws;               // [131072] floats
    float* partial = scores + M_TOTAL;           // [64*8*256] floats

    pass1_scores<<<M_TOTAL / MB, 256, 0, stream>>>(X, W, bvec, ctx, scores);
    pass2a_softmax<<<BATCH, 256, 0, stream>>>(scores);
    pass2b_partial<<<BATCH * NCHUNK, 256, 0, stream>>>(X, scores, partial);
    pass2c_reduce<<<BATCH, 256, 0, stream>>>(partial, out);
}
